// MultiLevelSimCLR_83777632076492
// MI455X (gfx1250) — compile-verified, matrix operand order checked by reading, not run on hardware
//
#include <hip/hip_runtime.h>
#include <hip/hip_bf16.h>
#include <math.h>

typedef __attribute__((ext_vector_type(16))) _Float16 v16h;
typedef __attribute__((ext_vector_type(8)))  float    v8f;

#define INS_B   8192
#define INS_N   16384      // 2*B
#define D_DIM   128
#define PATCH_B 256
#define PATCH_P 256
#define PATCH_N 512        // 2*P
#define INV_TEMP 10.0f
#define LSE_SHIFT 10.0f    // max possible logit = 1/TEMP (cosine sim in [-1,1])
#define LOG2E 1.4426950408889634f
// exp(c*INV_TEMP - LSE_SHIFT) == exp2(c*EXP2_SCALE + EXP2_BIAS).
// Argument range is [-28.9, 0] (cos in [-1,1]) -> far from the f32 underflow
// cliff, so the *raw* v_exp_f32 (no libm range fixup) is exact here.
#define EXP2_SCALE (INV_TEMP * LOG2E)
#define EXP2_BIAS  (-LSE_SHIFT * LOG2E)

__device__ __forceinline__ float fast_exp2(float x) {
  return __builtin_amdgcn_exp2f(x);      // bare v_exp_f32, no range guard
}

// ---------------------------------------------------------------------------
// WMMA fragment loader for Gram matrices (A and B patterns coincide).
// lanes 0-15 take halves [k0..k0+8) ++ [k0+16..k0+24); lanes 16-31 the rest.
// ---------------------------------------------------------------------------
__device__ __forceinline__ v16h load_frag(const _Float16* __restrict__ rowptr,
                                          int k0, int lane) {
  const _Float16* p = rowptr + k0 + ((lane & 16) ? 8 : 0);
  uint4 lo = *reinterpret_cast<const uint4*>(p);
  uint4 hi = *reinterpret_cast<const uint4*>(p + 16);
  v16h f;
  __builtin_memcpy(&f, &lo, 16);
  __builtin_memcpy(reinterpret_cast<char*>(&f) + 16, &hi, 16);
  return f;
}

__device__ __forceinline__ void load_tile(const _Float16* __restrict__ base,
                                          int j, int lrow, int lane,
                                          v16h& b0, v16h& b1, v16h& b2, v16h& b3) {
  const _Float16* r = base + (size_t)(j + lrow) * D_DIM;
  b0 = load_frag(r, 0, lane);
  b1 = load_frag(r, 32, lane);
  b2 = load_frag(r, 64, lane);
  b3 = load_frag(r, 96, lane);
}

__device__ __forceinline__ v8f wmma_f16(v16h a, v16h b, v8f c) {
  return __builtin_amdgcn_wmma_f32_16x16x32_f16(false, a, false, b,
                                                (short)0, c, false, false);
}

// Two 16-row groups against one shared 16-column B tile (K=128 -> 4 steps),
// interleaved so dependent accumulator chains alternate.
__device__ __forceinline__ void gemm2(const v16h* aA, const v16h* aB,
                                      const v16h& b0, const v16h& b1,
                                      const v16h& b2, const v16h& b3,
                                      v8f& cA, v8f& cB) {
  cA = wmma_f16(aA[0], b0, cA); cB = wmma_f16(aB[0], b0, cB);
  cA = wmma_f16(aA[1], b1, cA); cB = wmma_f16(aB[1], b1, cB);
  cA = wmma_f16(aA[2], b2, cA); cB = wmma_f16(aB[2], b2, cB);
  cA = wmma_f16(aA[3], b3, cA); cB = wmma_f16(aB[3], b3, cB);
}

__device__ __forceinline__ v8f gemm4(const v16h* a, const v16h& b0, const v16h& b1,
                                     const v16h& b2, const v16h& b3) {
  v8f c = {};
  c = wmma_f16(a[0], b0, c);
  c = wmma_f16(a[1], b1, c);
  c = wmma_f16(a[2], b2, c);
  c = wmma_f16(a[3], b3, c);
  return c;
}

// Branch-free accumulation: one pk_fma pair + one v_exp_f32 per element.
__device__ __forceinline__ void accum_plain(const v8f& c, float* __restrict__ sums) {
  #pragma unroll
  for (int v = 0; v < 8; ++v)
    sums[v] += fast_exp2(c[v] * EXP2_SCALE + EXP2_BIAS);
}

// ---------------------------------------------------------------------------
// Normalize instance embeddings: 16384 rows of 128 fp32 -> f16. 1 wave / row.
// ---------------------------------------------------------------------------
__global__ void __launch_bounds__(256)
normalize_ins_kernel(const float* __restrict__ v1, const float* __restrict__ v2,
                     _Float16* __restrict__ out) {
  const int lane = threadIdx.x & 31;
  const int wave = blockIdx.x * (blockDim.x >> 5) + (threadIdx.x >> 5);
  if (wave >= INS_N) return;
  const float* src = (wave < INS_B) ? (v1 + (size_t)wave * D_DIM)
                                    : (v2 + (size_t)(wave - INS_B) * D_DIM);
  float4 x = reinterpret_cast<const float4*>(src)[lane];
  float ss = x.x * x.x + x.y * x.y + x.z * x.z + x.w * x.w;
  #pragma unroll
  for (int m = 16; m >= 1; m >>= 1) ss += __shfl_xor(ss, m);
  float inv = 1.0f / fmaxf(sqrtf(ss), 1e-12f);
  union { _Float16 h[4]; uint2 u; } pk;
  pk.h[0] = (_Float16)(x.x * inv);
  pk.h[1] = (_Float16)(x.y * inv);
  pk.h[2] = (_Float16)(x.z * inv);
  pk.h[3] = (_Float16)(x.w * inv);
  *reinterpret_cast<uint2*>(out + (size_t)wave * D_DIM + lane * 4) = pk.u;
}

// ---------------------------------------------------------------------------
// Normalize patch embeddings: in (b, d, p) fp32, normalize over d, write f16
// transposed as (b, p', d).  Coalesced reads, 64KB LDS swizzled transpose.
// ---------------------------------------------------------------------------
__global__ void __launch_bounds__(256)
normalize_patch_kernel(const float* __restrict__ v1, const float* __restrict__ v2,
                       _Float16* __restrict__ out) {
  __shared__ _Float16 lds[PATCH_P * D_DIM];   // 65536 bytes
  const int b = blockIdx.x;
  const int src = blockIdx.y;
  const int p = threadIdx.x;
  const float* in = (src == 0 ? v1 : v2) + (size_t)b * D_DIM * PATCH_P;
  float ss = 0.f;
  for (int d = 0; d < D_DIM; ++d) {
    float x = in[(size_t)d * PATCH_P + p];
    ss += x * x;
  }
  float inv = 1.0f / fmaxf(sqrtf(ss), 1e-12f);
  for (int d = 0; d < D_DIM; ++d) {
    float x = in[(size_t)d * PATCH_P + p];
    lds[p * D_DIM + ((d + 8 * p) & 127)] = (_Float16)(x * inv);  // swizzled
  }
  __syncthreads();
  _Float16* dst = out + ((size_t)b * PATCH_N + (size_t)src * PATCH_P) * D_DIM;
  for (int c = p; c < (PATCH_P * D_DIM) / 8; c += 256) {   // 16B chunks
    int r  = c >> 4;
    int h0 = (c & 15) << 3;
    int hs = (h0 + 8 * r) & 127;                           // un-swizzle
    uint4 val = *reinterpret_cast<const uint4*>(&lds[r * D_DIM + hs]);
    *reinterpret_cast<uint4*>(dst + (size_t)c * 8) = val;
  }
}

// ---------------------------------------------------------------------------
// Instance loss: one wave per 32-row strip of the 16384x16384 Gram matrix.
// Branch-free hot loop (unrolled 2x, ping-pong B buffers redefined in place),
// diagonal/positive handled exactly by post-loop correction WMMAs.
// ---------------------------------------------------------------------------
__global__ void __launch_bounds__(128)
ins_loss_kernel(const _Float16* __restrict__ embs, float* __restrict__ partials) {
  const int lane = threadIdx.x & 31;
  const int wave =
      __builtin_amdgcn_readfirstlane(blockIdx.x * 4 + (threadIdx.x >> 5));
  const int row0  = wave << 5;            // 32 rows per wave
  const int rowgA = row0, rowgB = row0 + 16;
  const int labelA = (rowgA < INS_B) ? (rowgA + INS_B) : (rowgA - INS_B);
  const int lrow = lane & 15;
  const int hoff = (lane >> 4) << 3;

  v16h aA[4], aB[4];
  load_tile(embs, rowgA, lrow, lane, aA[0], aA[1], aA[2], aA[3]);
  load_tile(embs, rowgB, lrow, lane, aB[0], aB[1], aB[2], aB[3]);

  float sumsA[8], sumsB[8];
  #pragma unroll
  for (int v = 0; v < 8; ++v) { sumsA[v] = sumsB[v] = 0.f; }

  v16h P0, P1, P2, P3, Q0, Q1, Q2, Q3;
  load_tile(embs, 0,  lrow, lane, P0, P1, P2, P3);
  load_tile(embs, 16, lrow, lane, Q0, Q1, Q2, Q3);

  for (int j0 = 0; j0 < INS_N - 32; j0 += 32) {
    __builtin_prefetch(
        embs + (size_t)(((j0 + 64) & (INS_N - 1)) + lrow) * D_DIM, 0, 3);
    v8f cA = {}, cB = {};
    gemm2(aA, aB, P0, P1, P2, P3, cA, cB);
    load_tile(embs, j0 + 32, lrow, lane, P0, P1, P2, P3);   // redefine in place
    accum_plain(cA, sumsA);
    accum_plain(cB, sumsB);
    v8f dA = {}, dB = {};
    gemm2(aA, aB, Q0, Q1, Q2, Q3, dA, dB);
    load_tile(embs, j0 + 48, lrow, lane, Q0, Q1, Q2, Q3);
    accum_plain(dA, sumsA);
    accum_plain(dB, sumsB);
  }
  {   // peeled final two tiles (no further loads)
    v8f cA = {}, cB = {};
    gemm2(aA, aB, P0, P1, P2, P3, cA, cB);
    accum_plain(cA, sumsA);
    accum_plain(cB, sumsB);
    v8f dA = {}, dB = {};
    gemm2(aA, aB, Q0, Q1, Q2, Q3, dA, dB);
    accum_plain(dA, sumsA);
    accum_plain(dB, sumsB);
  }

  // ---- exact corrections ----
  // Diagonal tiles: B-fragments of a Gram diag tile are the A-fragments.
  v8f eA = gemm4(aA, aA[0], aA[1], aA[2], aA[3]);
  v8f eB = gemm4(aB, aB[0], aB[1], aB[2], aB[3]);
  // Positive tiles: label rows form the B-fragments.
  load_tile(embs, labelA,      lrow, lane, P0, P1, P2, P3);
  load_tile(embs, labelA + 16, lrow, lane, Q0, Q1, Q2, Q3);
  v8f pA = gemm4(aA, P0, P1, P2, P3);
  v8f pB = gemm4(aB, Q0, Q1, Q2, Q3);

  float posA[8], posB[8];
  #pragma unroll
  for (int v = 0; v < 8; ++v) {
    const bool own = (lrow == v + hoff);   // lane owning column r%16 of row r
    if (own) {
      sumsA[v] -= fast_exp2(eA[v] * EXP2_SCALE + EXP2_BIAS);  // remove diag term
      sumsB[v] -= fast_exp2(eB[v] * EXP2_SCALE + EXP2_BIAS);
      posA[v] = pA[v] * INV_TEMP;
      posB[v] = pB[v] * INV_TEMP;
    } else {
      posA[v] = 0.f;
      posB[v] = 0.f;
    }
  }

  #pragma unroll
  for (int v = 0; v < 8; ++v) {
    #pragma unroll
    for (int m = 1; m < 16; m <<= 1) {
      sumsA[v] += __shfl_xor(sumsA[v], m);
      posA[v]  += __shfl_xor(posA[v],  m);
      sumsB[v] += __shfl_xor(sumsB[v], m);
      posB[v]  += __shfl_xor(posB[v],  m);
    }
  }
  float acc = 0.f;
  #pragma unroll
  for (int v = 0; v < 8; ++v) {
    acc += (LSE_SHIFT + __logf(sumsA[v])) - posA[v];
    acc += (LSE_SHIFT + __logf(sumsB[v])) - posB[v];
  }
  acc += __shfl_xor(acc, 16);             // combine lane halves
  if (lane == 0) partials[wave] = acc;
}

// ---------------------------------------------------------------------------
// Patch loss: one wave per (batch, 32-row strip) of each 512x512 Gram.
// ---------------------------------------------------------------------------
__global__ void __launch_bounds__(128)
patch_loss_kernel(const _Float16* __restrict__ embs,
                  const int* __restrict__ c1, const int* __restrict__ c2,
                  float* __restrict__ partials) {
  const int lane = threadIdx.x & 31;
  const int wave =
      __builtin_amdgcn_readfirstlane(blockIdx.x * 4 + (threadIdx.x >> 5));
  const int b    = wave >> 4;            // 16 strips per batch
  const int row0 = (wave & 15) << 5;
  const int rowgA = row0, rowgB = row0 + 16;
  const int labelA = (rowgA < PATCH_P) ? (rowgA + PATCH_P) : (rowgA - PATCH_P);
  const int lrow = lane & 15;
  const int hoff = (lane >> 4) << 3;

  const _Float16* base = embs + (size_t)b * PATCH_N * D_DIM;
  v16h aA[4], aB[4];
  load_tile(base, rowgA, lrow, lane, aA[0], aA[1], aA[2], aA[3]);
  load_tile(base, rowgB, lrow, lane, aB[0], aB[1], aB[2], aB[3]);

  float sumsA[8], sumsB[8];
  #pragma unroll
  for (int v = 0; v < 8; ++v) { sumsA[v] = sumsB[v] = 0.f; }

  v16h P0, P1, P2, P3, Q0, Q1, Q2, Q3;
  load_tile(base, 0,  lrow, lane, P0, P1, P2, P3);
  load_tile(base, 16, lrow, lane, Q0, Q1, Q2, Q3);

  for (int j0 = 0; j0 < PATCH_N - 32; j0 += 32) {
    v8f cA = {}, cB = {};
    gemm2(aA, aB, P0, P1, P2, P3, cA, cB);
    load_tile(base, j0 + 32, lrow, lane, P0, P1, P2, P3);
    accum_plain(cA, sumsA);
    accum_plain(cB, sumsB);
    v8f dA = {}, dB = {};
    gemm2(aA, aB, Q0, Q1, Q2, Q3, dA, dB);
    load_tile(base, j0 + 48, lrow, lane, Q0, Q1, Q2, Q3);
    accum_plain(dA, sumsA);
    accum_plain(dB, sumsB);
  }
  {
    v8f cA = {}, cB = {};
    gemm2(aA, aB, P0, P1, P2, P3, cA, cB);
    accum_plain(cA, sumsA);
    accum_plain(cB, sumsB);
    v8f dA = {}, dB = {};
    gemm2(aA, aB, Q0, Q1, Q2, Q3, dA, dB);
    accum_plain(dA, sumsA);
    accum_plain(dB, sumsB);
  }

  v8f eA = gemm4(aA, aA[0], aA[1], aA[2], aA[3]);
  v8f eB = gemm4(aB, aB[0], aB[1], aB[2], aB[3]);
  load_tile(base, labelA,      lrow, lane, P0, P1, P2, P3);
  load_tile(base, labelA + 16, lrow, lane, Q0, Q1, Q2, Q3);
  v8f pA = gemm4(aA, P0, P1, P2, P3);
  v8f pB = gemm4(aB, Q0, Q1, Q2, Q3);

  float posA[8], posB[8];
  #pragma unroll
  for (int v = 0; v < 8; ++v) {
    const bool own = (lrow == v + hoff);
    if (own) {
      sumsA[v] -= fast_exp2(eA[v] * EXP2_SCALE + EXP2_BIAS);
      sumsB[v] -= fast_exp2(eB[v] * EXP2_SCALE + EXP2_BIAS);
      posA[v] = pA[v] * INV_TEMP;
      posB[v] = pB[v] * INV_TEMP;
    } else {
      posA[v] = 0.f;
      posB[v] = 0.f;
    }
  }

  #pragma unroll
  for (int v = 0; v < 8; ++v) {
    #pragma unroll
    for (int m = 1; m < 16; m <<= 1) {
      sumsA[v] += __shfl_xor(sumsA[v], m);
      posA[v]  += __shfl_xor(posA[v],  m);
      sumsB[v] += __shfl_xor(sumsB[v], m);
      posB[v]  += __shfl_xor(posB[v],  m);
    }
  }
  float acc = 0.f;
  #pragma unroll
  for (int v = 0; v < 8; ++v) {
    const int rA = rowgA + v + hoff;
    const int cntA = (rA < PATCH_P) ? c1[b * PATCH_P + rA]
                                    : c2[b * PATCH_P + rA - PATCH_P];
    if (cntA != 0) acc += (LSE_SHIFT + __logf(sumsA[v])) - posA[v];
    const int rB = rowgB + v + hoff;
    const int cntB = (rB < PATCH_P) ? c1[b * PATCH_P + rB]
                                    : c2[b * PATCH_P + rB - PATCH_P];
    if (cntB != 0) acc += (LSE_SHIFT + __logf(sumsB[v])) - posB[v];
  }
  acc += __shfl_xor(acc, 16);
  if (lane == 0) partials[wave] = acc;
}

// ---------------------------------------------------------------------------
// Deterministic fixed-order reduction of all partials -> scalar loss.
// ---------------------------------------------------------------------------
__global__ void __launch_bounds__(256)
finalize_kernel(const float* __restrict__ p_ins, const float* __restrict__ p_patch,
                const int* __restrict__ c1, const int* __restrict__ c2,
                float* __restrict__ out) {
  __shared__ float red[256];
  const int t = threadIdx.x;
  float s_ins = 0.f, s_pat = 0.f, s_nv = 0.f;
  for (int i = t; i < 512; i += 256) s_ins += p_ins[i];
  for (int i = t; i < 4096; i += 256) s_pat += p_patch[i];
  for (int i = t; i < PATCH_B * PATCH_P; i += 256) {
    s_nv += (c1[i] != 0) ? 1.0f : 0.0f;
    s_nv += (c2[i] != 0) ? 1.0f : 0.0f;
  }
  float tot_ins = 0.f, tot_pat = 0.f, tot_nv = 0.f;

  red[t] = s_ins; __syncthreads();
  for (int s = 128; s > 0; s >>= 1) { if (t < s) red[t] += red[t + s]; __syncthreads(); }
  tot_ins = red[0]; __syncthreads();

  red[t] = s_pat; __syncthreads();
  for (int s = 128; s > 0; s >>= 1) { if (t < s) red[t] += red[t + s]; __syncthreads(); }
  tot_pat = red[0]; __syncthreads();

  red[t] = s_nv; __syncthreads();
  for (int s = 128; s > 0; s >>= 1) { if (t < s) red[t] += red[t + s]; __syncthreads(); }
  tot_nv = red[0];

  if (t == 0) out[0] = tot_ins * (1.0f / (float)INS_N) + tot_pat / tot_nv;
}

// ---------------------------------------------------------------------------
extern "C" void kernel_launch(void* const* d_in, const int* in_sizes, int n_in,
                              void* d_out, int out_size, void* d_ws, size_t ws_size,
                              hipStream_t stream) {
  const float* v1_ins   = (const float*)d_in[0];
  const float* v2_ins   = (const float*)d_in[1];
  const float* v1_patch = (const float*)d_in[2];
  const float* v2_patch = (const float*)d_in[3];
  const int*   c1       = (const int*)d_in[4];
  const int*   c2       = (const int*)d_in[5];

  char* ws = (char*)d_ws;
  float*    p_ins      = (float*)ws;                                 // 512 floats
  float*    p_patch    = (float*)(ws + 4096);                        // 4096 floats
  _Float16* embs_ins   = (_Float16*)(ws + 36864);                    // 16384*128 f16 (4MB)
  _Float16* embs_patch = (_Float16*)(ws + 36864 + (size_t)INS_N * D_DIM * 2); // 32MB

  // 1) build f16 embedding tables
  normalize_ins_kernel<<<INS_N / 8, 256, 0, stream>>>(v1_ins, v2_ins, embs_ins);
  normalize_patch_kernel<<<dim3(PATCH_B, 2), 256, 0, stream>>>(v1_patch, v2_patch,
                                                               embs_patch);
  // 2) fused Gram + log-softmax reductions (WMMA f16 -> f32), 32 rows/wave
  ins_loss_kernel<<<(INS_N / 32) / 4, 128, 0, stream>>>(embs_ins, p_ins);
  patch_loss_kernel<<<(PATCH_B * (PATCH_N / 32)) / 4, 128, 0, stream>>>(
      embs_patch, c1, c2, p_patch);
  // 3) deterministic final reduction
  finalize_kernel<<<1, 256, 0, stream>>>(p_ins, p_patch, c1, c2, (float*)d_out);
}
